// spatial_attention_63960652972356
// MI455X (gfx1250) — matrix operand order repaired and checked by reading, not compiled, not run on hardware
//
#include <hip/hip_runtime.h>
#include <hip/hip_bf16.h>

// ---------------------------------------------------------------------------
// Shapes (compile-time constants from the reference)
// ---------------------------------------------------------------------------
#define S_LEN 64
#define P_NUM 1024
#define D_DIM 1024
#define H_DIM 512            // D/2
#define SP_ROWS (S_LEN * P_NUM)   // 65536

typedef __bf16 bhalf;
typedef __bf16 v4bf  __attribute__((ext_vector_type(4)));
typedef __bf16 v8bf  __attribute__((ext_vector_type(8)));
typedef __bf16 v16bf __attribute__((ext_vector_type(16)));
typedef float  v8f   __attribute__((ext_vector_type(8)));
typedef int    v4i   __attribute__((ext_vector_type(4)));

#define AS1 __attribute__((address_space(1)))
#define AS3 __attribute__((address_space(3)))

#if __has_builtin(__builtin_amdgcn_global_load_async_to_lds_b128) && \
    __has_builtin(__builtin_amdgcn_s_wait_asynccnt)
#define USE_ASYNC_LDS 1
#else
#define USE_ASYNC_LDS 0
#endif

// 16-byte chunk: global -> LDS (async DMA when available, else via one VGPR quad)
__device__ __forceinline__ void copy16_g2l(const bhalf* __restrict__ g,
                                           bhalf* __restrict__ l) {
#if USE_ASYNC_LDS
    __builtin_amdgcn_global_load_async_to_lds_b128((AS1 v4i*)g, (AS3 v4i*)l, 0, 0);
#else
    *(uint4*)l = *(const uint4*)g;
#endif
}

__device__ __forceinline__ void wait_async_copies() {
#if USE_ASYNC_LDS
    __builtin_amdgcn_s_wait_asynccnt(0);
#endif
}

// ---------------------------------------------------------------------------
// Weight convert + transpose: W [K,N] f32  ->  WT [N,K] bf16  (K contiguous)
// ---------------------------------------------------------------------------
__global__ void wconv_kernel(const float* __restrict__ W, bhalf* __restrict__ WT,
                             int K, int N) {
    int idx = blockIdx.x * 256 + threadIdx.x;   // grid sized exactly N*K/256
    int n = idx / K;
    int k = idx - n * K;
    WT[idx] = (bhalf)W[(size_t)k * N + n];
}

// ---------------------------------------------------------------------------
// x + pe -> bf16   (vectorized float4 in, 4x bf16 out)
// ---------------------------------------------------------------------------
__global__ void xpe_kernel(const float* __restrict__ x, const float* __restrict__ pe,
                           bhalf* __restrict__ out) {
    size_t i4 = (size_t)blockIdx.x * 256 + threadIdx.x;   // over S*P*D/4
    const size_t PE4 = (size_t)P_NUM * D_DIM / 4;         // 262144 (pow2)
    float4 xv = ((const float4*)x)[i4];
    float4 pv = ((const float4*)pe)[i4 & (PE4 - 1)];
    v4bf o;
    o[0] = (bhalf)(xv.x + pv.x);
    o[1] = (bhalf)(xv.y + pv.y);
    o[2] = (bhalf)(xv.z + pv.z);
    o[3] = (bhalf)(xv.w + pv.w);
    ((v4bf*)out)[i4] = o;
}

// ---------------------------------------------------------------------------
// Masked softmax over rows of 1024.
// ---------------------------------------------------------------------------
__global__ void softmax_mask_kernel(const float* __restrict__ logits,
                                    const float* __restrict__ dmask,
                                    float* __restrict__ outF,
                                    bhalf* __restrict__ outB) {
    __shared__ float red[8];
    const int tid = threadIdx.x;
    const size_t base = (size_t)blockIdx.x * P_NUM;

    float4 l = ((const float4*)(logits + base))[tid];
    float4 dm = ((const float4*)(dmask + base))[tid];
    l.x += (dm.x < 0.5f) ? -1e12f : 0.0f;
    l.y += (dm.y < 0.5f) ? -1e12f : 0.0f;
    l.z += (dm.z < 0.5f) ? -1e12f : 0.0f;
    l.w += (dm.w < 0.5f) ? -1e12f : 0.0f;

    float m = fmaxf(fmaxf(l.x, l.y), fmaxf(l.z, l.w));
    for (int o = 16; o > 0; o >>= 1) m = fmaxf(m, __shfl_xor(m, o, 32));
    if ((tid & 31) == 0) red[tid >> 5] = m;
    __syncthreads();
    float mall = red[0];
    #pragma unroll
    for (int i = 1; i < 8; ++i) mall = fmaxf(mall, red[i]);
    __syncthreads();

    float4 e;
    e.x = __expf(l.x - mall);
    e.y = __expf(l.y - mall);
    e.z = __expf(l.z - mall);
    e.w = __expf(l.w - mall);
    float s = e.x + e.y + e.z + e.w;
    for (int o = 16; o > 0; o >>= 1) s += __shfl_xor(s, o, 32);
    if ((tid & 31) == 0) red[tid >> 5] = s;
    __syncthreads();
    float sall = 0.0f;
    #pragma unroll
    for (int i = 0; i < 8; ++i) sall += red[i];
    float inv = 1.0f / sall;

    float4 r;
    r.x = e.x * inv; r.y = e.y * inv; r.z = e.z * inv; r.w = e.w * inv;
    ((float4*)(outF + base))[tid] = r;
    v4bf rb;
    rb[0] = (bhalf)r.x; rb[1] = (bhalf)r.y; rb[2] = (bhalf)r.z; rb[3] = (bhalf)r.w;
    ((v4bf*)(outB + base))[tid] = rb;
}

// ---------------------------------------------------------------------------
// Universal BF16 WMMA GEMM:  C[M,N] = A[M,K] @ BT[N,K]^T   (f32 accum)
// 128x128x64 tile, 8 waves (4x2), each wave = 2x4 grid of 16x16 WMMA tiles.
// Double-buffered LDS; tiles staged with GLOBAL_LOAD_ASYNC_TO_LDS_B128
// (zero VGPR footprint, tracked by ASYNCcnt).
// ---------------------------------------------------------------------------
#define BM 128
#define BN 128
#define BK 64
#define LDT 72   // padded LDS row stride (bf16): 144B -> conflict-free b128

__global__ __launch_bounds__(256)
__attribute__((amdgpu_waves_per_eu(1)))
void gemm_bf16_wmma_kernel(const bhalf* __restrict__ A, const bhalf* __restrict__ BT,
                           int M, int N, int K,
                           long long sA, long long sB, long long sC,
                           const float* __restrict__ bias, float alpha,
                           const float* __restrict__ gammaPtr,
                           const float* __restrict__ xres, const float* __restrict__ pe,
                           const float* __restrict__ addin,
                           float* __restrict__ outF, bhalf* __restrict__ outB,
                           int mode) {
    __shared__ __align__(16) bhalf sAt[2][BM * LDT];
    __shared__ __align__(16) bhalf sBt[2][BN * LDT];

    const int z = blockIdx.z;
    A  += (size_t)z * (size_t)sA;
    BT += (size_t)z * (size_t)sB;

    const int m0 = blockIdx.y * BM;
    const int n0 = blockIdx.x * BN;
    const int tid = threadIdx.x;
    const int lane = tid & 31;
    const int wave = tid >> 5;
    const int wm = wave & 3;   // 4 row-groups of 32
    const int wn = wave >> 2;  // 2 col-groups of 64

    // staging geometry: 1024 chunks of 16B cover a 128x64 bf16 tile
    const int srow = tid >> 3;          // rows 0..31 (+32 per step)
    const int sc8  = (tid & 7) << 3;    // bf16 offset within row

    auto stage_tile = [&](int k0, int buf) {
        #pragma unroll
        for (int it = 0; it < 4; ++it) {
            int row = srow + it * 32;
            copy16_g2l(A  + (size_t)(m0 + row) * K + k0 + sc8,
                       &sAt[buf][row * LDT + sc8]);
            copy16_g2l(BT + (size_t)(n0 + row) * K + k0 + sc8,
                       &sBt[buf][row * LDT + sc8]);
        }
    };

    v8f acc[2][4];
    #pragma unroll
    for (int mt = 0; mt < 2; ++mt)
        #pragma unroll
        for (int nt = 0; nt < 4; ++nt)
            acc[mt][nt] = (v8f)(0.0f);

    stage_tile(0, 0);
    wait_async_copies();
    __syncthreads();

    const int KT = K / BK;
    const int arow  = (wm << 5) + (lane & 15);
    const int akoff = (lane >> 4) << 3;   // 0 or 8   (ISA 16-bit A layout)
    const int brow  = (wn << 6) + (lane & 15);
    const int bkoff = (lane >> 4) << 3;   // 0 or 8   (same k order as A)

    for (int kt = 0; kt < KT; ++kt) {
        const int cur = kt & 1;
        if (kt + 1 < KT) stage_tile((kt + 1) * BK, cur ^ 1);  // async into other buffer

        #pragma unroll
        for (int kk = 0; kk < 2; ++kk) {   // two K=32 WMMA steps per LDS tile
            const int kb = kk * 32;
            v16bf af[2], bfrag[4];
            #pragma unroll
            for (int mt = 0; mt < 2; ++mt) {
                const bhalf* p = &sAt[cur][(arow + mt * 16) * LDT + kb + akoff];
                v8bf lo = *(const v8bf*)p;            // K = kb+akoff .. +7
                v8bf hi = *(const v8bf*)(p + 16);     // K = kb+16+akoff .. +7
                af[mt] = __builtin_shufflevector(lo, hi, 0, 1, 2, 3, 4, 5, 6, 7,
                                                 8, 9, 10, 11, 12, 13, 14, 15);
            }
            #pragma unroll
            for (int nt = 0; nt < 4; ++nt) {
                const bhalf* p = &sBt[cur][(brow + nt * 16) * LDT + kb + bkoff];
                v8bf lo = *(const v8bf*)p;            // K = kb+bkoff .. +7
                v8bf hi = *(const v8bf*)(p + 16);     // K = kb+16+bkoff .. +7
                bfrag[nt] = __builtin_shufflevector(lo, hi, 0, 1, 2, 3, 4, 5, 6, 7,
                                                    8, 9, 10, 11, 12, 13, 14, 15);
            }
            #pragma unroll
            for (int mt = 0; mt < 2; ++mt)
                #pragma unroll
                for (int nt = 0; nt < 4; ++nt)
                    acc[mt][nt] = __builtin_amdgcn_wmma_f32_16x16x32_bf16(
                        false, af[mt], false, bfrag[nt], (short)0, acc[mt][nt],
                        false, false);
        }

        wait_async_copies();   // this wave's async stores into buf^1 are done
        __syncthreads();       // everyone finished reading cur + staging cur^1
    }

    // ----------------------------- epilogue --------------------------------
    const float gval = gammaPtr ? gammaPtr[0] : 0.0f;
    #pragma unroll
    for (int mt = 0; mt < 2; ++mt) {
        #pragma unroll
        for (int nt = 0; nt < 4; ++nt) {
            const int nn = n0 + (wn << 6) + nt * 16 + (lane & 15);
            const float bv = bias ? bias[nn] : 0.0f;
            const int mbase = m0 + (wm << 5) + mt * 16 + ((lane >> 4) << 3);
            v8f c = acc[mt][nt];
            #pragma unroll
            for (int r = 0; r < 8; ++r) {
                const int mm = mbase + r;
                const float cv = c[r];
                const size_t idx = (size_t)mm * N + nn;
                switch (mode) {
                case 0: outB[idx] = (bhalf)(cv + bv); break;
                case 1: outB[idx] = (bhalf)((cv + bv) * alpha); break;
                case 2: outB[(size_t)z * sC + (size_t)nn * M + mm] = (bhalf)(cv + bv); break;
                case 3: outF[(size_t)z * sC + idx] = cv; break;
                case 4: {
                    const size_t o = (size_t)z * sC + idx;
                    const float res = xres[o] + pe[idx];
                    const float v = gval * cv + res;
                    outF[o] = v;
                    outB[o] = (bhalf)v;
                    break;
                }
                case 5: outB[idx] = (bhalf)fmaxf(cv + bv, 0.0f); break;
                case 6: outF[idx] = cv + bv + addin[idx]; break;
                }
            }
        }
    }
}

// ---------------------------------------------------------------------------
// Host-side orchestration
// ---------------------------------------------------------------------------
extern "C" void kernel_launch(void* const* d_in, const int* in_sizes, int n_in,
                              void* d_out, int out_size, void* d_ws, size_t ws_size,
                              hipStream_t stream) {
    (void)in_sizes; (void)n_in; (void)out_size; (void)ws_size;

    const float* x     = (const float*)d_in[0];
    const float* dmask = (const float*)d_in[1];
    const float* pe    = (const float*)d_in[2];
    const float* Wq    = (const float*)d_in[3];
    const float* bq    = (const float*)d_in[4];
    const float* Wk    = (const float*)d_in[5];
    const float* bk    = (const float*)d_in[6];
    const float* Wv    = (const float*)d_in[7];
    const float* bv    = (const float*)d_in[8];
    const float* gamma = (const float*)d_in[9];
    const float* W1    = (const float*)d_in[10];
    const float* b1    = (const float*)d_in[11];
    const float* W2    = (const float*)d_in[12];
    const float* b2    = (const float*)d_in[13];
    const float* W3    = (const float*)d_in[14];
    const float* b3    = (const float*)d_in[15];

    // ---- workspace layout (bytes) ----
    const size_t SZ_SPD_BF = (size_t)SP_ROWS * D_DIM * 2;   // 128 MB
    const size_t SZ_SPP_F  = (size_t)SP_ROWS * P_NUM * 4;   // 256 MB
    char* ws = (char*)d_ws;
    bhalf* xpe    = (bhalf*)(ws);                                   // 128 MB
    bhalf* qb     = (bhalf*)(ws + SZ_SPD_BF);                       // 128 MB
    bhalf* kb     = (bhalf*)(ws + 2 * SZ_SPD_BF);                   // 128 MB
    bhalf* vT     = (bhalf*)(ws + 3 * SZ_SPD_BF);                   // 128 MB
    float* logits = (float*)(ws + 4 * SZ_SPD_BF);                   // 256 MB
    bhalf* amap   = (bhalf*)(ws + 4 * SZ_SPD_BF + SZ_SPP_F);        // 128 MB
    char*  wb     = ws + 5 * SZ_SPD_BF + SZ_SPP_F;
    bhalf* WqT = (bhalf*)(wb);                 // D*D bf16 = 2 MB
    bhalf* WkT = (bhalf*)(wb + 2097152);
    bhalf* WvT = (bhalf*)(wb + 2 * 2097152);
    bhalf* W1T = (bhalf*)(wb + 3 * 2097152);               // H*D = 1 MB
    bhalf* W2T = (bhalf*)(wb + 3 * 2097152 + 1048576);     // H*H = 0.5 MB
    bhalf* W3T = (bhalf*)(wb + 3 * 2097152 + 1572864);     // D*H = 1 MB
    // reuse dead regions:
    bhalf* h1   = (bhalf*)(ws);                        // over xpe (dead)
    bhalf* h2   = (bhalf*)(ws + (size_t)SP_ROWS * H_DIM * 2);
    bhalf* outb = qb;                                   // over q (dead)
    float* outf = logits;                               // over logits (dead)

    float* y_out    = (float*)d_out;
    float* attn_out = (float*)d_out + (size_t)SP_ROWS * D_DIM;

    // ---- 1) weight convert + transpose ----
    wconv_kernel<<<(D_DIM * D_DIM) / 256, 256, 0, stream>>>(Wq, WqT, D_DIM, D_DIM);
    wconv_kernel<<<(D_DIM * D_DIM) / 256, 256, 0, stream>>>(Wk, WkT, D_DIM, D_DIM);
    wconv_kernel<<<(D_DIM * D_DIM) / 256, 256, 0, stream>>>(Wv, WvT, D_DIM, D_DIM);
    wconv_kernel<<<(H_DIM * D_DIM) / 256, 256, 0, stream>>>(W1, W1T, D_DIM, H_DIM);
    wconv_kernel<<<(H_DIM * H_DIM) / 256, 256, 0, stream>>>(W2, W2T, H_DIM, H_DIM);
    wconv_kernel<<<(D_DIM * H_DIM) / 256, 256, 0, stream>>>(W3, W3T, H_DIM, D_DIM);

    // ---- 2) x + pe -> bf16 ----
    xpe_kernel<<<((size_t)SP_ROWS * D_DIM / 4) / 256, 256, 0, stream>>>(x, pe, xpe);

    const long long PD = (long long)P_NUM * D_DIM;
    const long long PP = (long long)P_NUM * P_NUM;

    // ---- 3) q = (xpe @ WqT^T + bq) * (1/sqrt(P)) ----
    gemm_bf16_wmma_kernel<<<dim3(D_DIM / BN, SP_ROWS / BM, 1), 256, 0, stream>>>(
        xpe, WqT, SP_ROWS, D_DIM, D_DIM, 0, 0, 0,
        bq, 0.03125f, nullptr, nullptr, nullptr, nullptr, nullptr, qb, 1);
    // ---- 4) k ----
    gemm_bf16_wmma_kernel<<<dim3(D_DIM / BN, SP_ROWS / BM, 1), 256, 0, stream>>>(
        xpe, WkT, SP_ROWS, D_DIM, D_DIM, 0, 0, 0,
        bk, 1.0f, nullptr, nullptr, nullptr, nullptr, nullptr, kb, 0);
    // ---- 5) v (stored transposed per batch: vT[s][d][p]) ----
    gemm_bf16_wmma_kernel<<<dim3(D_DIM / BN, P_NUM / BM, S_LEN), 256, 0, stream>>>(
        xpe, WvT, P_NUM, D_DIM, D_DIM, PD, 0, PD,
        bv, 1.0f, nullptr, nullptr, nullptr, nullptr, nullptr, vT, 2);
    // ---- 6) attn logits: q @ k^T (batched over s) ----
    gemm_bf16_wmma_kernel<<<dim3(P_NUM / BN, P_NUM / BM, S_LEN), 256, 0, stream>>>(
        qb, kb, P_NUM, P_NUM, D_DIM, PD, PD, PP,
        nullptr, 1.0f, nullptr, nullptr, nullptr, nullptr, logits, nullptr, 3);
    // ---- 7) masked softmax -> attn_map (f32 output) + bf16 copy ----
    softmax_mask_kernel<<<SP_ROWS, 256, 0, stream>>>(logits, dmask, attn_out, amap);
    // ---- 8) out = gamma * (attn_map @ v) + (x + pe)  -> f32 + bf16 ----
    gemm_bf16_wmma_kernel<<<dim3(D_DIM / BN, P_NUM / BM, S_LEN), 256, 0, stream>>>(
        amap, vT, P_NUM, D_DIM, P_NUM, PP, PD, PD,
        nullptr, 1.0f, gamma, x, pe, nullptr, outf, outb, 4);
    // ---- 9) h1 = relu(out @ W1T^T + b1) ----
    gemm_bf16_wmma_kernel<<<dim3(H_DIM / BN, SP_ROWS / BM, 1), 256, 0, stream>>>(
        outb, W1T, SP_ROWS, H_DIM, D_DIM, 0, 0, 0,
        b1, 1.0f, nullptr, nullptr, nullptr, nullptr, nullptr, h1, 5);
    // ---- 10) h2 = relu(h1 @ W2T^T + b2) ----
    gemm_bf16_wmma_kernel<<<dim3(H_DIM / BN, SP_ROWS / BM, 1), 256, 0, stream>>>(
        h1, W2T, SP_ROWS, H_DIM, H_DIM, 0, 0, 0,
        b2, 1.0f, nullptr, nullptr, nullptr, nullptr, nullptr, h2, 5);
    // ---- 11) y = h2 @ W3T^T + b3 + out ----
    gemm_bf16_wmma_kernel<<<dim3(D_DIM / BN, SP_ROWS / BM, 1), 256, 0, stream>>>(
        h2, W3T, SP_ROWS, D_DIM, H_DIM, 0, 0, 0,
        b3, 1.0f, nullptr, nullptr, nullptr, outf, y_out, nullptr, 6);
}
